// GNN_71236327572210
// MI455X (gfx1250) — matrix operand order repaired and checked by reading, not compiled, not run on hardware
//
#include <hip/hip_runtime.h>
#include <hip/hip_bf16.h>
#include <math.h>

typedef __attribute__((ext_vector_type(16))) __bf16 v16bf;
typedef __attribute__((ext_vector_type(8)))  __bf16 bf16x8;
typedef __attribute__((ext_vector_type(8)))  float  v8f;

#define NODES   100000
#define DEG     16
#define IN_F    128
#define HID_F   256
#define CLS_F   40
#define CLS_PAD 64
#define MTILE   32
#define BLOCK   256

// ---------- prep kernels ----------

__global__ void cvt_f32_bf16(const float* __restrict__ x, __bf16* __restrict__ y, int n) {
  int i = blockIdx.x * blockDim.x + threadIdx.x;
  if (i < n) y[i] = (__bf16)x[i];
}

// Wt[n][k] = W[k][n] (bf16), zero-padded columns for n >= Nreal
__global__ void transpose_w(const float* __restrict__ W, __bf16* __restrict__ Wt,
                            int K, int Nreal, int Npad) {
  int i = blockIdx.x * blockDim.x + threadIdx.x;
  if (i >= Npad * K) return;
  int n = i / K;
  int k = i - n * K;
  float v = (n < Nreal) ? W[(size_t)k * Nreal + n] : 0.0f;
  Wt[i] = (__bf16)v;
}

__global__ void pad_bias(const float* __restrict__ b, float* __restrict__ bp,
                         int nreal, int npad) {
  int i = blockIdx.x * blockDim.x + threadIdx.x;
  if (i < npad) bp[i] = (i < nreal) ? b[i] : 0.0f;
}

// ---------- fused aggregate + GEMM layer ----------
// agg = (h[self] + sum_{e<16} h[src[node*16+e]]) / 16   (dst = repeat(arange(N),16))
// out = act(agg @ W + b), activations kept in bf16, accumulation in f32 via WMMA.

template<int K, int NOUT, bool RELU, bool OUTF32>
__global__ __launch_bounds__(BLOCK) void sage_layer(
    const __bf16* __restrict__ hin, const __bf16* __restrict__ Wt,
    const float* __restrict__ bias, const int* __restrict__ src,
    __bf16* __restrict__ outB, float* __restrict__ outF) {

  constexpr int WT = NOUT / 64;   // column tiles (of 16) per wave, 4 waves across N
  constexpr int KP = K + 8;       // LDS row padding (16B) to spread banks

  __shared__ int sSrc[MTILE * DEG];
  __shared__ __align__(16) __bf16 sAgg[MTILE][KP];

  const int tid   = threadIdx.x;
  const int node0 = blockIdx.x * MTILE;

  for (int i = tid; i < MTILE * DEG; i += BLOCK)
    sSrc[i] = src[(size_t)node0 * DEG + i];
  __syncthreads();

  // ---- phase 1: gather + mean-aggregate into LDS (bf16) ----
  constexpr int CPR = K / 8;                       // 16B chunks per node row
  for (int c = tid; c < MTILE * CPR; c += BLOCK) {
    int m  = c / CPR;
    int kb = (c - m * CPR) * 8;
    float acc[8];
    bf16x8 v = *(const bf16x8*)(hin + (size_t)(node0 + m) * K + kb);
    #pragma unroll
    for (int i = 0; i < 8; ++i) acc[i] = (float)v[i];
    #pragma unroll
    for (int e = 0; e < DEG; ++e) {
      size_t s = (size_t)sSrc[m * DEG + e];
      bf16x8 u = *(const bf16x8*)(hin + s * K + kb);
      #pragma unroll
      for (int i = 0; i < 8; ++i) acc[i] += (float)u[i];
    }
    bf16x8 o;
    #pragma unroll
    for (int i = 0; i < 8; ++i) o[i] = (__bf16)(acc[i] * 0.0625f);
    *(bf16x8*)&sAgg[m][kb] = o;
  }
  __syncthreads();

  // ---- phase 2: WMMA GEMM, 8 waves as 2 (M) x 4 (N) ----
  const int wid  = tid >> 5;
  const int lane = tid & 31;
  const int mw   = wid >> 2;             // 0..1  -> 16-row tile
  const int nw   = wid & 3;              // 0..3  -> NOUT/4-wide strip
  const int colbase = nw * (NOUT / 4);
  const int r  = lane & 15;
  const int hi = lane >> 4;

  v8f acc[WT] = {};

  for (int kk = 0; kk < K; kk += 32) {
    // A fragment (16x32 bf16): lane<16 holds K {kk..kk+7, kk+16..kk+23} of row M=lane,
    // lane>=16 holds K {kk+8..kk+15, kk+24..kk+31} of row M=lane-16.
    const __bf16* ap = &sAgg[mw * 16 + r][kk + hi * 8];
    bf16x8 alo  = *(const bf16x8*)ap;
    bf16x8 ahi8 = *(const bf16x8*)(ap + 16);
    v16bf a;
    #pragma unroll
    for (int i = 0; i < 8; ++i) { a[i] = alo[i]; a[i + 8] = ahi8[i]; }

    #pragma unroll
    for (int t = 0; t < WT; ++t) {
      // B fragment (32x16 bf16): lane holds column N=lane%16,
      // K {kk+8*hi..+7, kk+16+8*hi..+7} (same order as A) from Wt[n][k].
      const __bf16* bp = Wt + (size_t)(colbase + t * 16 + r) * K + kk + hi * 8;
      bf16x8 blo = *(const bf16x8*)bp;
      bf16x8 bhi8 = *(const bf16x8*)(bp + 16);
      v16bf b;
      #pragma unroll
      for (int i = 0; i < 8; ++i) { b[i] = blo[i]; b[i + 8] = bhi8[i]; }
      acc[t] = __builtin_amdgcn_wmma_f32_16x16x32_bf16(
          false, a, false, b, (short)0, acc[t], false, false);
    }
  }

  // ---- epilogue: bias (+relu), store. C layout: M = v + 8*hi, N = lane%16 ----
  const int rbase = node0 + mw * 16 + hi * 8;
  #pragma unroll
  for (int t = 0; t < WT; ++t) {
    const int   ncol = colbase + t * 16 + r;
    const float bv   = bias[ncol];
    #pragma unroll
    for (int v = 0; v < 8; ++v) {
      float x = acc[t][v] + bv;
      if (RELU) x = fmaxf(x, 0.0f);
      const size_t idx = (size_t)(rbase + v) * NOUT + ncol;
      if (OUTF32) outF[idx] = x;
      else        outB[idx] = (__bf16)x;
    }
  }
}

// ---------- row-wise log_softmax over the 40 valid logits ----------

__global__ void log_softmax_k(const float* __restrict__ logits, float* __restrict__ out) {
  int row = blockIdx.x * blockDim.x + threadIdx.x;
  if (row >= NODES) return;
  const float* p = logits + (size_t)row * CLS_PAD;
  float v[CLS_F];
  float mx = -3.4e38f;
  #pragma unroll
  for (int c = 0; c < CLS_F; ++c) { v[c] = p[c]; mx = fmaxf(mx, v[c]); }
  float s = 0.0f;
  #pragma unroll
  for (int c = 0; c < CLS_F; ++c) s += __expf(v[c] - mx);
  float lse = mx + __logf(s);
  float* q = out + (size_t)row * CLS_F;
  #pragma unroll
  for (int c = 0; c < CLS_F; ++c) q[c] = v[c] - lse;
}

// ---------- launch ----------

extern "C" void kernel_launch(void* const* d_in, const int* in_sizes, int n_in,
                              void* d_out, int out_size, void* d_ws, size_t ws_size,
                              hipStream_t stream) {
  (void)in_sizes; (void)n_in; (void)out_size; (void)ws_size;

  const float* features = (const float*)d_in[0];
  const float* W1 = (const float*)d_in[1];
  const float* b1 = (const float*)d_in[2];
  const float* W2 = (const float*)d_in[3];
  const float* b2 = (const float*)d_in[4];
  const float* W3 = (const float*)d_in[5];
  const float* b3 = (const float*)d_in[6];
  const int*  src = (const int*)d_in[7];   // harness passes integer inputs as int32
  // d_in[8] = dst is implicit: repeat(arange(N), DEG), in-degree == DEG

  char* ws = (char*)d_ws;
  const size_t off_h0 = 0;
  const size_t off_h1 = (size_t)NODES * IN_F  * 2;           // 25.6 MB
  const size_t off_h2 = off_h1 + (size_t)NODES * HID_F * 2;  // +51.2 MB
  const size_t off_w1 = off_h2 + (size_t)NODES * HID_F * 2;  // +51.2 MB
  const size_t off_w2 = off_w1 + (size_t)HID_F * IN_F  * 2;
  const size_t off_w3 = off_w2 + (size_t)HID_F * HID_F * 2;
  const size_t off_b3 = off_w3 + (size_t)CLS_PAD * HID_F * 2;

  __bf16* h0     = (__bf16*)(ws + off_h0);
  float*  logits = (float*)(ws + off_h0);   // aliases h0 (h0 dead after layer 1; same size)
  __bf16* h1     = (__bf16*)(ws + off_h1);
  __bf16* h2     = (__bf16*)(ws + off_h2);
  __bf16* Wt1    = (__bf16*)(ws + off_w1);
  __bf16* Wt2    = (__bf16*)(ws + off_w2);
  __bf16* Wt3    = (__bf16*)(ws + off_w3);
  float*  b3p    = (float*)(ws + off_b3);

  cvt_f32_bf16<<<(NODES * IN_F + 255) / 256, 256, 0, stream>>>(features, h0, NODES * IN_F);
  transpose_w<<<(HID_F * IN_F + 255) / 256, 256, 0, stream>>>(W1, Wt1, IN_F, HID_F, HID_F);
  transpose_w<<<(HID_F * HID_F + 255) / 256, 256, 0, stream>>>(W2, Wt2, HID_F, HID_F, HID_F);
  transpose_w<<<(CLS_PAD * HID_F + 255) / 256, 256, 0, stream>>>(W3, Wt3, HID_F, CLS_F, CLS_PAD);
  pad_bias<<<1, 64, 0, stream>>>(b3, b3p, CLS_F, CLS_PAD);

  const int blocks = NODES / MTILE;  // 100000 / 32 = 3125, exact
  sage_layer<IN_F,  HID_F,  true,  false><<<blocks, BLOCK, 0, stream>>>(h0, Wt1, b1,  src, h1, nullptr);
  sage_layer<HID_F, HID_F,  true,  false><<<blocks, BLOCK, 0, stream>>>(h1, Wt2, b2,  src, h2, nullptr);
  sage_layer<HID_F, CLS_PAD, false, true><<<blocks, BLOCK, 0, stream>>>(h2, Wt3, b3p, src, nullptr, logits);

  log_softmax_k<<<(NODES + 255) / 256, 256, 0, stream>>>(logits, (float*)d_out);
}
